// TransformerLayer_43748536877701
// MI455X (gfx1250) — matrix operand order repaired and checked by reading, not compiled, not run on hardware
//
#include <hip/hip_runtime.h>
#include <hip/hip_bf16.h>

// ---------------------------------------------------------------------------
// Problem constants (from reference)
// ---------------------------------------------------------------------------
#define LSEQ   2048
#define DIM    512
#define HEADS  8
#define DHEAD  64
#define KS     127
#define NRPB   (2 * KS - 1)   // 253
#define DFF    (DIM * 4)      // 2048

typedef __attribute__((ext_vector_type(16))) __bf16 v16bf;
typedef __attribute__((ext_vector_type(8)))  __bf16 v8bf;
typedef __attribute__((ext_vector_type(8)))  float  v8f;

static __device__ __forceinline__ v16bf mk16(v8bf lo, v8bf hi) {
    v16bf r;
#pragma unroll
    for (int i = 0; i < 8; ++i) { r[i] = lo[i]; r[i + 8] = hi[i]; }
    return r;
}

// ---------------------------------------------------------------------------
// Weight prep: fp32 W[K,N]  ->  bf16 Wt[N,K]   (tiled transpose-convert)
// ---------------------------------------------------------------------------
__global__ __launch_bounds__(256) void k_transpose_bf16(const float* __restrict__ W,
                                                        __bf16* __restrict__ Wt,
                                                        int K, int N) {
    __shared__ float tile[32][33];
    const int n0 = blockIdx.x * 32;
    const int k0 = blockIdx.y * 32;
    const int tx = threadIdx.x;   // 0..31
    const int ty = threadIdx.y;   // 0..7
#pragma unroll
    for (int i = 0; i < 32; i += 8)
        tile[ty + i][tx] = W[(size_t)(k0 + ty + i) * N + (n0 + tx)];
    __syncthreads();
#pragma unroll
    for (int i = 0; i < 32; i += 8)
        Wt[(size_t)(n0 + ty + i) * K + (k0 + tx)] = (__bf16)tile[tx][ty + i];
}

// ---------------------------------------------------------------------------
// LayerNorm over DIM=512, one block (256 threads) per row, bf16 output
// ---------------------------------------------------------------------------
__global__ __launch_bounds__(256) void k_layernorm(const float* __restrict__ x,
                                                   const float* __restrict__ g,
                                                   const float* __restrict__ b,
                                                   __bf16* __restrict__ out) {
    __shared__ float red[256];
    const int row = blockIdx.x;
    const int tid = threadIdx.x;
    const float* xr = x + (size_t)row * DIM;

    const float v0 = xr[tid];
    const float v1 = xr[tid + 256];

    red[tid] = v0 + v1;
    __syncthreads();
#pragma unroll
    for (int off = 128; off > 0; off >>= 1) {
        if (tid < off) red[tid] += red[tid + off];
        __syncthreads();
    }
    const float mean = red[0] * (1.0f / DIM);
    __syncthreads();

    const float d0 = v0 - mean, d1 = v1 - mean;
    red[tid] = d0 * d0 + d1 * d1;
    __syncthreads();
#pragma unroll
    for (int off = 128; off > 0; off >>= 1) {
        if (tid < off) red[tid] += red[tid + off];
        __syncthreads();
    }
    const float rstd = rsqrtf(red[0] * (1.0f / DIM) + 1e-5f);

    __bf16* o = out + (size_t)row * DIM;
    o[tid]       = (__bf16)(d0 * rstd * g[tid]       + b[tid]);
    o[tid + 256] = (__bf16)(d1 * rstd * g[tid + 256] + b[tid + 256]);
}

// ---------------------------------------------------------------------------
// WMMA GEMM: C[M,N] = A[M,K](bf16,row-major) x Bt[N,K](bf16) + bias, epilogue
// MODE 0: outF = acc + bias
// MODE 1: outF = acc + bias + res            (residual add)
// MODE 2: outB = bf16(gelu(acc + bias))      (exact gelu)
// Block tile 128x64, BK=64, 256 threads = 8 waves, each wave a 32x32 sub-tile.
// ---------------------------------------------------------------------------
template <int MODE>
__global__ __launch_bounds__(256) void k_gemm_wmma(const __bf16* __restrict__ A,
                                                   const __bf16* __restrict__ Bt,
                                                   const float* __restrict__ bias,
                                                   const float* __restrict__ res,
                                                   float* __restrict__ outF,
                                                   __bf16* __restrict__ outB,
                                                   int M, int N, int K) {
    constexpr int BM = 128, BN = 64, BK = 64, PAD = 8;
    constexpr int LDA = BK + PAD;   // halves
    constexpr int LDB = BK + PAD;
    __shared__ __align__(16) __bf16 As[BM * LDA];
    __shared__ __align__(16) __bf16 Bs[BN * LDB];

    const int tid  = threadIdx.x;
    const int m0   = blockIdx.y * BM;
    const int n0   = blockIdx.x * BN;
    const int wv   = tid >> 5;
    const int lane = tid & 31;
    const int wm   = (wv & 3) * 32;   // wave row offset in tile
    const int wn   = (wv >> 2) * 32;  // wave col offset in tile
    const int lr   = lane & 15;
    const int lh   = lane >> 4;

    v8f c[2][2] = {};

    for (int k0 = 0; k0 < K; k0 += BK) {
        // ---- global -> LDS (16B chunks; all tiles divide evenly) ----
#pragma unroll
        for (int i = 0; i < 4; ++i) {                     // A: 128 rows x 128B
            const int ch  = tid + i * 256;
            const int row = ch >> 3;
            const int col = (ch & 7) * 8;                 // halves
            *(uint4*)(&As[row * LDA + col]) =
                *(const uint4*)(A + (size_t)(m0 + row) * K + k0 + col);
        }
#pragma unroll
        for (int i = 0; i < 2; ++i) {                     // B: 64 rows x 128B
            const int ch  = tid + i * 256;
            const int row = ch >> 3;
            const int col = (ch & 7) * 8;
            *(uint4*)(&Bs[row * LDB + col]) =
                *(const uint4*)(Bt + (size_t)(n0 + row) * K + k0 + col);
        }
        __syncthreads();

        // ---- two K=32 WMMA steps per BK ----
#pragma unroll
        for (int kk = 0; kk < 2; ++kk) {
            v16bf a[2], b[2];
#pragma unroll
            for (int mi = 0; mi < 2; ++mi) {
                const __bf16* p = &As[(wm + mi * 16 + lr) * LDA + kk * 32 + lh * 8];
                a[mi] = mk16(*(const v8bf*)p, *(const v8bf*)(p + 16));
            }
#pragma unroll
            for (int ni = 0; ni < 2; ++ni) {
                const __bf16* p = &Bs[(wn + ni * 16 + lr) * LDB + kk * 32 + lh * 8];
                b[ni] = mk16(*(const v8bf*)p, *(const v8bf*)(p + 16));
            }
#pragma unroll
            for (int mi = 0; mi < 2; ++mi)
#pragma unroll
                for (int ni = 0; ni < 2; ++ni)
                    c[mi][ni] = __builtin_amdgcn_wmma_f32_16x16x32_bf16(
                        false, a[mi], false, b[ni], (short)0, c[mi][ni], false, false);
        }
        __syncthreads();
    }

    // ---- epilogue: C layout VGPR r -> (M = r + (lane>>4)*8, N = lane&15) ----
#pragma unroll
    for (int mi = 0; mi < 2; ++mi)
#pragma unroll
        for (int ni = 0; ni < 2; ++ni) {
            const int gmb = m0 + wm + mi * 16 + lh * 8;
            const int gn  = n0 + wn + ni * 16 + lr;
            const float bv = bias[gn];
#pragma unroll
            for (int r = 0; r < 8; ++r) {
                const int row = gmb + r;
                float v = c[mi][ni][r] + bv;
                if constexpr (MODE == 1) v += res[(size_t)row * N + gn];
                if constexpr (MODE == 2) {
                    const float t = 0.5f * v * (1.0f + erff(v * 0.70710678f));
                    outB[(size_t)row * N + gn] = (__bf16)t;
                } else {
                    outF[(size_t)row * N + gn] = v;
                }
            }
        }
}

// ---------------------------------------------------------------------------
// Neighborhood attention (1D, window 127, clipped starts).
// One block per position l (256 threads); wave w handles head w.
// qkv layout per row (1536 f32): [q | k | v], each [HEADS][DHEAD].
// ---------------------------------------------------------------------------
__global__ __launch_bounds__(256) void k_natten(const float* __restrict__ qkv,
                                                const float* __restrict__ rpb,
                                                __bf16* __restrict__ out) {
    __shared__ float qs[HEADS][DHEAD];
    __shared__ float at[HEADS][KS + 1];

    const int l    = blockIdx.x;
    const int wv   = threadIdx.x >> 5;   // head
    const int lane = threadIdx.x & 31;

    int start = l - (KS / 2);
    if (start < 0) start = 0;
    if (start > LSEQ - KS) start = LSEQ - KS;

    const float scale = 0.125f;          // 1/sqrt(64)
    const float* qrow = qkv + (size_t)l * (3 * DIM) + wv * DHEAD;
    qs[wv][lane]      = qrow[lane]      * scale;
    qs[wv][lane + 32] = qrow[lane + 32] * scale;
    __syncthreads();

    // logits: 4 keys per lane
    float lg[4];
#pragma unroll
    for (int j = 0; j < 4; ++j) {
        const int kk = lane + j * 32;
        float acc = -1e30f;
        if (kk < KS) {
            const float* krow = qkv + (size_t)(start + kk) * (3 * DIM) + DIM + wv * DHEAD;
            float s = 0.0f;
#pragma unroll
            for (int d = 0; d < DHEAD; d += 4) {
                const float4 kq = *(const float4*)(krow + d);
                const float4 qq = *(const float4*)(&qs[wv][d]);
                s += qq.x * kq.x + qq.y * kq.y + qq.z * kq.z + qq.w * kq.w;
            }
            s += rpb[wv * NRPB + (start + kk - l + (KS - 1))];
            acc = s;
        }
        lg[j] = acc;
    }

    // softmax over the 127 logits spread across the wave
    float mx = fmaxf(fmaxf(lg[0], lg[1]), fmaxf(lg[2], lg[3]));
#pragma unroll
    for (int off = 16; off > 0; off >>= 1) mx = fmaxf(mx, __shfl_xor(mx, off, 32));
    float e[4], sum = 0.0f;
#pragma unroll
    for (int j = 0; j < 4; ++j) {
        e[j] = (lg[j] > -1e29f) ? __expf(lg[j] - mx) : 0.0f;
        sum += e[j];
    }
#pragma unroll
    for (int off = 16; off > 0; off >>= 1) sum += __shfl_xor(sum, off, 32);
    const float inv = 1.0f / sum;
#pragma unroll
    for (int j = 0; j < 4; ++j) {
        const int kk = lane + j * 32;
        if (kk < KS) at[wv][kk] = e[j] * inv;
    }
    __syncthreads();

    // weighted sum of V: lane handles 2 consecutive dims (coalesced per key)
    const int d0 = lane * 2;
    float a0 = 0.0f, a1 = 0.0f;
    const float* vbase = qkv + (size_t)start * (3 * DIM) + 2 * DIM + wv * DHEAD + d0;
    for (int kk = 0; kk < KS; ++kk) {
        const float w = at[wv][kk];
        const float* vp = vbase + (size_t)kk * (3 * DIM);
        a0 += w * vp[0];
        a1 += w * vp[1];
    }
    __bf16* o = out + (size_t)l * DIM + wv * DHEAD + d0;
    o[0] = (__bf16)a0;
    o[1] = (__bf16)a1;
}

// ---------------------------------------------------------------------------
// Host-side orchestration
// ---------------------------------------------------------------------------
extern "C" void kernel_launch(void* const* d_in, const int* in_sizes, int n_in,
                              void* d_out, int out_size, void* d_ws, size_t ws_size,
                              hipStream_t stream) {
    (void)in_sizes; (void)n_in; (void)out_size; (void)ws_size;

    const float* x      = (const float*)d_in[0];
    const float* ln1_g  = (const float*)d_in[1];
    const float* ln1_b  = (const float*)d_in[2];
    const float* qkv_w  = (const float*)d_in[3];
    const float* qkv_b  = (const float*)d_in[4];
    const float* rpb    = (const float*)d_in[5];
    const float* proj_w = (const float*)d_in[6];
    const float* proj_b = (const float*)d_in[7];
    const float* ln2_g  = (const float*)d_in[8];
    const float* ln2_b  = (const float*)d_in[9];
    const float* ff1_w  = (const float*)d_in[10];
    const float* ff1_b  = (const float*)d_in[11];
    const float* ff2_w  = (const float*)d_in[12];
    const float* ff2_b  = (const float*)d_in[13];
    float* outp = (float*)d_out;

    char* ws = (char*)d_ws;
    size_t off = 0;
    auto alloc = [&](size_t bytes) -> char* {
        char* p = ws + off;
        off += (bytes + 255) & ~(size_t)255;
        return p;
    };
    __bf16* qkvWt  = (__bf16*)alloc((size_t)(3 * DIM) * DIM * 2);   // [1536,512]
    __bf16* projWt = (__bf16*)alloc((size_t)DIM * DIM * 2);         // [512,512]
    __bf16* ff1Wt  = (__bf16*)alloc((size_t)DFF * DIM * 2);         // [2048,512]
    __bf16* ff2Wt  = (__bf16*)alloc((size_t)DIM * DFF * 2);         // [512,2048]
    __bf16* lnbuf  = (__bf16*)alloc((size_t)LSEQ * DIM * 2);        // ln1/ln2 out
    float*  qkvf   = (float*) alloc((size_t)LSEQ * 3 * DIM * 4);    // qkv f32
    __bf16* attno  = (__bf16*)alloc((size_t)LSEQ * DIM * 2);        // attn out bf16
    float*  x2     = (float*) alloc((size_t)LSEQ * DIM * 4);        // x + attn proj
    __bf16* h1     = (__bf16*)alloc((size_t)LSEQ * DFF * 2);        // gelu(ff1)

    const dim3 tb(32, 8);
    // Weight transpose+convert: W[K,N] -> Wt[N,K] bf16
    k_transpose_bf16<<<dim3((3 * DIM) / 32, DIM / 32), tb, 0, stream>>>(qkv_w, qkvWt, DIM, 3 * DIM);
    k_transpose_bf16<<<dim3(DIM / 32, DIM / 32),       tb, 0, stream>>>(proj_w, projWt, DIM, DIM);
    k_transpose_bf16<<<dim3(DFF / 32, DIM / 32),       tb, 0, stream>>>(ff1_w, ff1Wt, DIM, DFF);
    k_transpose_bf16<<<dim3(DIM / 32, DFF / 32),       tb, 0, stream>>>(ff2_w, ff2Wt, DFF, DIM);

    // ln1
    k_layernorm<<<LSEQ, 256, 0, stream>>>(x, ln1_g, ln1_b, lnbuf);
    // qkv = ln1 @ qkv_w + qkv_b          (MODE 0, f32 out)
    k_gemm_wmma<0><<<dim3((3 * DIM) / 64, LSEQ / 128), 256, 0, stream>>>(
        lnbuf, qkvWt, qkv_b, nullptr, qkvf, nullptr, LSEQ, 3 * DIM, DIM);
    // neighborhood attention -> bf16
    k_natten<<<LSEQ, 256, 0, stream>>>(qkvf, rpb, attno);
    // x2 = x + attn @ proj_w + proj_b    (MODE 1, residual x)
    k_gemm_wmma<1><<<dim3(DIM / 64, LSEQ / 128), 256, 0, stream>>>(
        attno, projWt, proj_b, x, x2, nullptr, LSEQ, DIM, DIM);
    // ln2
    k_layernorm<<<LSEQ, 256, 0, stream>>>(x2, ln2_g, ln2_b, lnbuf);
    // h1 = gelu(ln2 @ ff1_w + ff1_b)     (MODE 2, bf16 out)
    k_gemm_wmma<2><<<dim3(DFF / 64, LSEQ / 128), 256, 0, stream>>>(
        lnbuf, ff1Wt, ff1_b, nullptr, nullptr, h1, LSEQ, DFF, DIM);
    // out = x2 + h1 @ ff2_w + ff2_b      (MODE 1, residual x2)
    k_gemm_wmma<1><<<dim3(DIM / 64, LSEQ / 128), 256, 0, stream>>>(
        h1, ff2Wt, ff2_b, x2, outp, nullptr, LSEQ, DIM, DFF);
}
